// DeadlockGNN_74560632258655
// MI455X (gfx1250) — matrix operand order repaired and checked by reading, not compiled, not run on hardware
//
#include <hip/hip_runtime.h>

typedef __attribute__((ext_vector_type(16))) _Float16 v16h;
typedef __attribute__((ext_vector_type(8)))  float    v8f;

#define DH 64

// ---------------------------------------------------------------- utilities
__global__ void zero_f32(float* __restrict__ p, long n) {
  long i = (long)blockIdx.x * blockDim.x + threadIdx.x;
  if (i < n) p[i] = 0.0f;
}

__global__ void edge_count(const int* __restrict__ dst, float* __restrict__ cnt, int E) {
  int e = blockIdx.x * blockDim.x + threadIdx.x;
  if (e < E) atomicAdd(&cnt[dst[e]], 1.0f);
}

__global__ void make_inv(const float* __restrict__ cnt, float* __restrict__ inv, int n) {
  int i = blockIdx.x * blockDim.x + threadIdx.x;
  if (i < n) inv[i] = 1.0f / fmaxf(cnt[i], 1.0f);
}

// ---------------------------------------------- weight pre-pack (f32 -> f16)
// Packs W (DIN x 64 row-major, K x N) into per-lane WMMA B-fragments:
//   P[((kc*4 + nt)*32 + lane)*16 + e],
//   k = kc*32 + (e>>3)*16 + (lane>>4)*8 + (e&7), n = nt*16 + (lane&15)
template <int DIN>
__global__ void pack_w(const float* __restrict__ W, _Float16* __restrict__ P) {
  constexpr int KC = (DIN + 31) / 32;
  int idx = blockIdx.x * blockDim.x + threadIdx.x;
  if (idx >= KC * 4 * 32 * 16) return;
  int e    = idx & 15;
  int lane = (idx >> 4) & 31;
  int nt   = (idx >> 9) & 3;
  int kc   = idx >> 11;
  int k = kc * 32 + (e >> 3) * 16 + (lane >> 4) * 8 + (e & 7);
  int n = nt * 16 + (lane & 15);
  P[idx] = (_Float16)((k < DIN) ? W[(long)k * DH + n] : 0.0f);
}

// -------------------------------------------------- edge gather/scatter-add
template <int D>
__global__ void scatter_add4(const float* __restrict__ h,
                             const int* __restrict__ src,
                             const int* __restrict__ dst,
                             float* __restrict__ agg, int E) {
  constexpr int Q = D / 4;
  long idx = (long)blockIdx.x * blockDim.x + threadIdx.x;
  if (idx >= (long)E * Q) return;
  int e = (int)(idx / Q);
  int j = (int)(idx % Q);
  int s = src[e];
  int d = dst[e];
  const float4 v = ((const float4*)(h + (long)s * D))[j];
  float* dp = agg + (long)d * D + j * 4;
  atomicAdd(dp + 0, v.x);
  atomicAdd(dp + 1, v.y);
  atomicAdd(dp + 2, v.z);
  atomicAdd(dp + 3, v.w);
}

// ------------------------------------------------------- SAGE layer (WMMA)
// out[n,:] = act( mean[n,:] @ Wl + b + x[n,:] @ Wr ),  mean = agg * inv_cnt
// One wave per 16-node tile; weights staged in LDS as pre-packed fragments.
// POOL: instead of storing, atomicAdd rows into gpool[batch[row]] (layer 3).
// REQUIRES: N % 16 == 0 (true: 100000 = 6250*16), so no per-row guards.
template <int DIN, bool RELU, bool POOL>
__global__ __launch_bounds__(256) void sage_wmma(
    const float* __restrict__ x,       // N x DIN
    const float* __restrict__ agg,     // N x DIN
    const float* __restrict__ invc,    // N
    const _Float16* __restrict__ Pl,   // packed Wl fragments
    const _Float16* __restrict__ Pr,   // packed Wr fragments
    const float* __restrict__ bias,    // 64
    float* __restrict__ out,           // N x 64 (unused if POOL)
    const int* __restrict__ batch,     // N      (used if POOL)
    float* __restrict__ gpool,         // G x 64 (used if POOL)
    int N) {
  constexpr int KC = (DIN + 31) / 32;
  constexpr int FR = KC * 4 * 32 * 16;          // halfs per packed matrix
  __shared__ __align__(32) _Float16 sWl[FR];
  __shared__ __align__(32) _Float16 sWr[FR];

  {  // cooperative LDS fill: 16B chunks
    const uint4* gl = (const uint4*)Pl;
    const uint4* gr = (const uint4*)Pr;
    uint4* ll = (uint4*)sWl;
    uint4* lr = (uint4*)sWr;
    for (int i = threadIdx.x; i < FR / 8; i += 256) {
      ll[i] = gl[i];
      lr[i] = gr[i];
    }
  }
  __syncthreads();

  const int lane = threadIdx.x & 31;
  const int wave = threadIdx.x >> 5;
  const int tile = (blockIdx.x * 8 + wave) * 16;
  if (tile >= N) return;  // uniform per wave -> EXEC all-ones at every WMMA

  const int h2 = lane >> 4;
  const int node = tile + (lane & 15);
  const float scale = invc[node];
  const float* xr = x   + (long)node * DIN;
  const float* ar = agg + (long)node * DIN;

  // A fragments (16-bit A 16x32 layout). Per 32-K chunk, a lane needs two
  // contiguous 8-float runs: elems 0..7 <- row[kc*32 + h2*8 + 0..7],
  // elems 8..15 <- row[kc*32 + 16 + h2*8 + 0..7]. Branch-free: clamp the
  // offset into valid memory and fold the zero-pad into a multiplier.
  v16h aM[KC], aX[KC];
#pragma unroll
  for (int kc = 0; kc < KC; ++kc) {
#pragma unroll
    for (int half = 0; half < 2; ++half) {
      const int b = kc * 32 + half * 16 + h2 * 8;
      float mm, mx;
      int o;
      if (DIN >= 32) {           // compile-time: always in range
        o = b; mm = scale; mx = 1.0f;
      } else {                   // DIN=8: only b==0 is valid
        const bool ok = (b + 8 <= DIN);
        o = ok ? b : 0;
        mm = ok ? scale : 0.0f;
        mx = ok ? 1.0f : 0.0f;
      }
      const float4 a0 = *(const float4*)(ar + o);
      const float4 a1 = *(const float4*)(ar + o + 4);
      const float4 x0 = *(const float4*)(xr + o);
      const float4 x1 = *(const float4*)(xr + o + 4);
      const int eb = half * 8;
      aM[kc][eb + 0] = (_Float16)(a0.x * mm);
      aM[kc][eb + 1] = (_Float16)(a0.y * mm);
      aM[kc][eb + 2] = (_Float16)(a0.z * mm);
      aM[kc][eb + 3] = (_Float16)(a0.w * mm);
      aM[kc][eb + 4] = (_Float16)(a1.x * mm);
      aM[kc][eb + 5] = (_Float16)(a1.y * mm);
      aM[kc][eb + 6] = (_Float16)(a1.z * mm);
      aM[kc][eb + 7] = (_Float16)(a1.w * mm);
      aX[kc][eb + 0] = (_Float16)(x0.x * mx);
      aX[kc][eb + 1] = (_Float16)(x0.y * mx);
      aX[kc][eb + 2] = (_Float16)(x0.z * mx);
      aX[kc][eb + 3] = (_Float16)(x0.w * mx);
      aX[kc][eb + 4] = (_Float16)(x1.x * mx);
      aX[kc][eb + 5] = (_Float16)(x1.y * mx);
      aX[kc][eb + 6] = (_Float16)(x1.z * mx);
      aX[kc][eb + 7] = (_Float16)(x1.w * mx);
    }
  }

  // Pool path: batch ids for this lane's 8 output rows, loaded once.
  int bid[8];
  if (POOL) {
#pragma unroll
    for (int r = 0; r < 8; ++r) bid[r] = batch[tile + h2 * 8 + r];
  }

  const v16h* fl = (const v16h*)sWl;
  const v16h* fr = (const v16h*)sWr;
  const int nl = lane & 15;
#pragma unroll
  for (int nt = 0; nt < 4; ++nt) {
    const int ncol = nt * 16 + nl;
    v8f acc = {};
#pragma unroll
    for (int kc = 0; kc < KC; ++kc) {
      v16h bl = fl[(kc * 4 + nt) * 32 + lane];   // ds_load_b128 x2
      v16h br = fr[(kc * 4 + nt) * 32 + lane];
      acc = __builtin_amdgcn_wmma_f32_16x16x32_f16(false, aM[kc], false, bl,
                                                   (short)0, acc, false, false);
      acc = __builtin_amdgcn_wmma_f32_16x16x32_f16(false, aX[kc], false, br,
                                                   (short)0, acc, false, false);
    }
    const float bb = bias[ncol];
    if (POOL) {
#pragma unroll
      for (int r = 0; r < 8; ++r)
        atomicAdd(&gpool[(long)bid[r] * DH + ncol], acc[r] + bb);
    } else {
      float* op = out + (long)(tile + h2 * 8) * DH + ncol;
#pragma unroll
      for (int r = 0; r < 8; ++r) {
        // C/D f32 layout: VGPR r -> M = r + 8*h2, N = lane&15
        float v = acc[r] + bb;
        if (RELU) v = fmaxf(v, 0.0f);
        op[(long)r * DH] = v;
      }
    }
  }
}

// ------------------------------------------------------------------- head
__global__ void head_kernel(const float* __restrict__ g, const float* __restrict__ Wc1,
                            const float* __restrict__ bc1, const float* __restrict__ Wc2,
                            const float* __restrict__ bc2, float* __restrict__ out, int G) {
  int gi = blockIdx.x * blockDim.x + threadIdx.x;
  if (gi >= G) return;
  const float* gv = g + (long)gi * DH;
  float o = bc2[0];
#pragma unroll 4
  for (int j = 0; j < 32; ++j) {
    float hsum = bc1[j];
#pragma unroll 8
    for (int k = 0; k < DH; ++k) hsum += gv[k] * Wc1[k * 32 + j];
    o += fmaxf(hsum, 0.0f) * Wc2[j];
  }
  out[gi] = o;
}

// ------------------------------------------------------------------ launch
static inline int gsz(long n, int b) { return (int)((n + b - 1) / b); }

extern "C" void kernel_launch(void* const* d_in, const int* in_sizes, int n_in,
                              void* d_out, int out_size, void* d_ws, size_t ws_size,
                              hipStream_t stream) {
  const int N = 100000, E = 1200000, G = 256;  // N % 16 == 0 (sage_wmma relies on it)

  const float* x   = (const float*)d_in[0];
  const int*   ei  = (const int*)d_in[1];
  const int*   bat = (const int*)d_in[2];
  const float* W1l = (const float*)d_in[3];
  const float* b1  = (const float*)d_in[4];
  const float* W1r = (const float*)d_in[5];
  const float* W2l = (const float*)d_in[6];
  const float* b2  = (const float*)d_in[7];
  const float* W2r = (const float*)d_in[8];
  const float* W3l = (const float*)d_in[9];
  const float* b3  = (const float*)d_in[10];
  const float* W3r = (const float*)d_in[11];
  const float* Wc1 = (const float*)d_in[12];
  const float* bc1 = (const float*)d_in[13];
  const float* Wc2 = (const float*)d_in[14];
  const float* bc2 = (const float*)d_in[15];
  float* out = (float*)d_out;

  const int* src = ei;
  const int* dst = ei + E;

  // workspace carve-up (all offsets 16B-aligned)
  float* ws    = (float*)d_ws;
  float* cnt   = ws;                       // N
  float* invc  = cnt + N;                  // N
  float* agg   = invc + N;                 // N*64 (layer 1 uses N*8 of it)
  float* hA    = agg + (long)N * DH;       // N*64
  float* hB    = hA + (long)N * DH;        // N*64
  float* gpool = hB + (long)N * DH;        // 256*64
  _Float16* P1l = (_Float16*)(gpool + (long)G * DH);  // 2048 halfs
  _Float16* P1r = P1l + 2048;
  _Float16* P2l = P1r + 2048;              // 4096 halfs each below
  _Float16* P2r = P2l + 4096;
  _Float16* P3l = P2r + 4096;
  _Float16* P3r = P3l + 4096;

  const int B = 256;
  const int gemmBlocks = ((N + 15) / 16 + 7) / 8;

  // pre-pack all weights into WMMA fragment order (tiny, once per launch)
  pack_w<8><<<gsz(2048, B), B, 0, stream>>>(W1l, P1l);
  pack_w<8><<<gsz(2048, B), B, 0, stream>>>(W1r, P1r);
  pack_w<DH><<<gsz(4096, B), B, 0, stream>>>(W2l, P2l);
  pack_w<DH><<<gsz(4096, B), B, 0, stream>>>(W2r, P2r);
  pack_w<DH><<<gsz(4096, B), B, 0, stream>>>(W3l, P3l);
  pack_w<DH><<<gsz(4096, B), B, 0, stream>>>(W3r, P3r);

  // degree counts (shared across all three layers)
  zero_f32<<<gsz(2L * N, B), B, 0, stream>>>(cnt, 2L * N);
  edge_count<<<gsz(E, B), B, 0, stream>>>(dst, cnt, E);
  make_inv<<<gsz(N, B), B, 0, stream>>>(cnt, invc, N);

  // ---- layer 1: D_IN = 8 -> 64, relu
  zero_f32<<<gsz((long)N * 8, B), B, 0, stream>>>(agg, (long)N * 8);
  scatter_add4<8><<<gsz((long)E * 2, B), B, 0, stream>>>(x, src, dst, agg, E);
  sage_wmma<8, true, false><<<gemmBlocks, B, 0, stream>>>(
      x, agg, invc, P1l, P1r, b1, hA, nullptr, nullptr, N);

  // ---- layer 2: 64 -> 64, relu
  zero_f32<<<gsz((long)N * DH, B), B, 0, stream>>>(agg, (long)N * DH);
  scatter_add4<DH><<<gsz((long)E * 16, B), B, 0, stream>>>(hA, src, dst, agg, E);
  sage_wmma<DH, true, false><<<gemmBlocks, B, 0, stream>>>(
      hA, agg, invc, P2l, P2r, b2, hB, nullptr, nullptr, N);

  // ---- layer 3: 64 -> 64, no relu, fused global_add_pool into epilogue
  zero_f32<<<gsz((long)N * DH, B), B, 0, stream>>>(agg, (long)N * DH);
  scatter_add4<DH><<<gsz((long)E * 16, B), B, 0, stream>>>(hB, src, dst, agg, E);
  zero_f32<<<gsz((long)G * DH, B), B, 0, stream>>>(gpool, (long)G * DH);
  sage_wmma<DH, false, true><<<gemmBlocks, B, 0, stream>>>(
      hB, agg, invc, P3l, P3r, b3, nullptr, bat, gpool, N);

  // ---- MLP head
  head_kernel<<<1, 256, 0, stream>>>(gpool, Wc1, bc1, Wc2, bc2, out, G);
}
